// EnhancedSelfCalibratingAttention_87754771792452
// MI455X (gfx1250) — matrix operand order repaired and checked by reading, not compiled, not run on hardware
//
#include <hip/hip_runtime.h>
#include <hip/hip_bf16.h>
#include <math.h>

// ---------------------------------------------------------------------------
// Problem constants (from the reference)
// ---------------------------------------------------------------------------
#define DIMV   1024
#define HDIM   512          // dim/2 (scorer hidden)
#define QHID   256          // dim/4 (quant-net hidden)
#define SEQV   4096
#define BATCHV 4
#define KTOK   2048         // NUM_TOKENS
#define LN_EPS 1e-5f

typedef __attribute__((ext_vector_type(16))) _Float16 v16h;
typedef __attribute__((ext_vector_type(8)))  float    v8f;

// ---------------------------------------------------------------------------
// Kernel 0: convert w1 (1024 x 512, f32, row-major K-major) to f16, laid out
// exactly in WMMA B-fragment order:
//   dst[((kc*32 + ntile)*32 + lane)*16 + e]
//   where kc = k>>5, kk = k&31, ntile = n>>4,
//         lane = ((kk>>3)&1)*16 + (n&15), e = (kk&7) | ((kk>>4)<<3)
// so a wave's B fragment for (kc, ntile) is 32 lanes x 32 contiguous bytes.
// 1 MB total -> stays L2-resident for the whole scorer.
// ---------------------------------------------------------------------------
__global__ __launch_bounds__(256)
void w1_swizzle_kernel(const float* __restrict__ w1, _Float16* __restrict__ w1sw)
{
    const int i = blockIdx.x * 256 + threadIdx.x;     // over 1024*512
    const int k  = i >> 9, n = i & (HDIM - 1);
    const int kc = k >> 5, kk = k & 31;
    const int lane = (((kk >> 3) & 1) << 4) | (n & 15);
    const int e    = (kk & 7) | ((kk >> 4) << 3);
    const int ntg  = n >> 4;
    w1sw[(((size_t)kc * 32 + ntg) * 32 + lane) * 16 + e] = (_Float16)w1[i];
}

// ---------------------------------------------------------------------------
// Kernel 1: importance scorer.
//   h = relu(LN(x @ w1 + b1)); imp = sigmoid(h @ w2 + b2)
// Block = 16 tokens (1 M-tile) x 512 cols, 256 threads = 8 waves (wave32).
// Wave w owns N-tiles [4w, 4w+4) -> acc = 4 x v8f kept in ONE register set
// (K-loop not unrolled -> no accumulator rotation copies).
// A fragment: straight from global x (two aligned float4 runs per lane).
// B fragments: 4x (2x global_load_b128) from pre-swizzled L2-resident w1sw,
// loaded into explicit registers, then 4 independent WMMAs back-to-back.
// ---------------------------------------------------------------------------
__global__ __launch_bounds__(256)
void scorer_kernel(const float* __restrict__ x, const _Float16* __restrict__ w1sw,
                   const float* __restrict__ b1, const float* __restrict__ g1,
                   const float* __restrict__ beta1,
                   const float* __restrict__ w2, const float* __restrict__ b2,
                   float* __restrict__ imp)
{
    __shared__ float rs[16], rs2[16], rsc[16];

    const int tid  = threadIdx.x;
    const int wave = tid >> 5;         // N-group: 4 tiles = 64 cols
    const int lane = tid & 31;
    const int hi   = lane >> 4;        // WMMA lane half
    const int lo   = lane & 15;

    const int b  = blockIdx.x >> 8;            // / (SEQV/16)
    const int t0 = (blockIdx.x & 255) * 16;

    if (tid < 16) { rs[tid] = 0.f; rs2[tid] = 0.f; rsc[tid] = 0.f; }

    // this lane's A row and K-phase: K(e) = hi*8 + (e<8 ? e : e+8)
    const int arow = t0 + lo;
    const float* xr = x + ((size_t)b * SEQV + arow) * DIMV + hi * 8;

    v8f acc0 = (v8f){0.f,0.f,0.f,0.f,0.f,0.f,0.f,0.f};
    v8f acc1 = acc0, acc2 = acc0, acc3 = acc0;

#pragma unroll 1
    for (int kc = 0; kc < DIMV / 32; ++kc) {
        // ---- A fragment: two contiguous runs of 8 f32, convert to f16 ----
        const float4* ap = (const float4*)(xr + kc * 32);
        const float4 f0 = ap[0], f1 = ap[1];   // K = hi*8 + 0..7
        const float4 f2 = ap[4], f3 = ap[5];   // K = hi*8 + 16..23

        // ---- B fragments: 4 tiles, 32B per lane each ----
        const _Float16* wb = w1sw + (((size_t)kc * 32 + wave * 4) * 32 + lane) * 16;
        const v16h bm0 = *(const v16h*)(wb);
        const v16h bm1 = *(const v16h*)(wb + 512);
        const v16h bm2 = *(const v16h*)(wb + 1024);
        const v16h bm3 = *(const v16h*)(wb + 1536);
        if (kc + 1 < DIMV / 32)
            __builtin_prefetch(wb + (size_t)32 * 32 * 16, 0, 1);

        v16h a;
        a[0]=(_Float16)f0.x; a[1]=(_Float16)f0.y; a[2]=(_Float16)f0.z; a[3]=(_Float16)f0.w;
        a[4]=(_Float16)f1.x; a[5]=(_Float16)f1.y; a[6]=(_Float16)f1.z; a[7]=(_Float16)f1.w;
        a[8]=(_Float16)f2.x; a[9]=(_Float16)f2.y; a[10]=(_Float16)f2.z; a[11]=(_Float16)f2.w;
        a[12]=(_Float16)f3.x; a[13]=(_Float16)f3.y; a[14]=(_Float16)f3.z; a[15]=(_Float16)f3.w;

        // 4 independent WMMAs (disjoint D, shared A) -> no hazard NOPs
        acc0 = __builtin_amdgcn_wmma_f32_16x16x32_f16(false, a, false, bm0, (short)0, acc0, false, false);
        acc1 = __builtin_amdgcn_wmma_f32_16x16x32_f16(false, a, false, bm1, (short)0, acc1, false, false);
        acc2 = __builtin_amdgcn_wmma_f32_16x16x32_f16(false, a, false, bm2, (short)0, acc2, false, false);
        acc3 = __builtin_amdgcn_wmma_f32_16x16x32_f16(false, a, false, bm3, (short)0, acc3, false, false);
    }

    // per-column constants for this lane's 4 tiles
    float b1c[4], g1c[4], be1c[4], w2c[4];
#pragma unroll
    for (int nt = 0; nt < 4; ++nt) {
        const int col = wave * 64 + nt * 16 + lo;
        b1c[nt] = b1[col]; g1c[nt] = g1[col]; be1c[nt] = beta1[col]; w2c[nt] = w2[col];
    }

    __syncthreads();   // rs/rs2/rsc zeroed

    // C layout: VGPR r, lane l -> row = (l>=16)*8 + r, col = l&15
    float vals[4][8];
#pragma unroll
    for (int r = 0; r < 8; ++r) {
        vals[0][r] = acc0[r] + b1c[0];
        vals[1][r] = acc1[r] + b1c[1];
        vals[2][r] = acc2[r] + b1c[2];
        vals[3][r] = acc3[r] + b1c[3];
    }

    // bias + per-lane partial LN sums over the 4 tiles, then 16-lane butterfly
    float p1[8], p2[8];
#pragma unroll
    for (int r = 0; r < 8; ++r) { p1[r] = 0.f; p2[r] = 0.f; }
#pragma unroll
    for (int nt = 0; nt < 4; ++nt)
#pragma unroll
        for (int r = 0; r < 8; ++r) {
            const float v = vals[nt][r];
            p1[r] += v;
            p2[r] += v * v;
        }
#pragma unroll
    for (int r = 0; r < 8; ++r) {
#pragma unroll
        for (int m = 1; m < 16; m <<= 1) {       // reduce over lo group, keep hi
            p1[r] += __shfl_xor(p1[r], m, 32);
            p2[r] += __shfl_xor(p2[r], m, 32);
        }
        if (lo == 0) {
            const int rowl = hi * 8 + r;
            atomicAdd(&rs[rowl],  p1[r]);
            atomicAdd(&rs2[rowl], p2[r]);
        }
    }
    __syncthreads();

    float mu[8], rstd[8];
#pragma unroll
    for (int r = 0; r < 8; ++r) {
        const int rowl = hi * 8 + r;
        const float m = rs[rowl] * (1.f / HDIM);
        mu[r] = m;
        const float var = rs2[rowl] * (1.f / HDIM) - m * m;
        rstd[r] = rsqrtf(var + LN_EPS);
    }

    float ps[8];
#pragma unroll
    for (int r = 0; r < 8; ++r) ps[r] = 0.f;
#pragma unroll
    for (int nt = 0; nt < 4; ++nt)
#pragma unroll
        for (int r = 0; r < 8; ++r) {
            float h = (vals[nt][r] - mu[r]) * rstd[r] * g1c[nt] + be1c[nt];
            h = fmaxf(h, 0.f);
            ps[r] += h * w2c[nt];
        }
#pragma unroll
    for (int r = 0; r < 8; ++r) {
#pragma unroll
        for (int m = 1; m < 16; m <<= 1)
            ps[r] += __shfl_xor(ps[r], m, 32);
        if (lo == 0)
            atomicAdd(&rsc[hi * 8 + r], ps[r]);
    }
    __syncthreads();

    if (tid < 16) {
        const float s = rsc[tid] + b2[0];
        imp[(size_t)b * SEQV + t0 + tid] = 1.f / (1.f + __expf(-s));
    }
}

// ---------------------------------------------------------------------------
// Kernel 2: per-batch full bitonic sort (descending by score, index tiebreak)
// of 4096 (score, idx) pairs in LDS; emit first KTOK indices = top-k order.
// ---------------------------------------------------------------------------
__global__ __launch_bounds__(1024)
void topk_kernel(const float* __restrict__ imp, int* __restrict__ idxout)
{
    __shared__ float v[SEQV];
    __shared__ int   ix[SEQV];
    const int b = blockIdx.x;

    for (int i = threadIdx.x; i < SEQV; i += 1024) {
        v[i]  = imp[(size_t)b * SEQV + i];
        ix[i] = i;
    }
    __syncthreads();

    for (int k = 2; k <= SEQV; k <<= 1) {
        for (int j = k >> 1; j > 0; j >>= 1) {
            for (int i = threadIdx.x; i < SEQV; i += 1024) {
                const int p = i ^ j;
                if (p > i) {
                    const bool desc = ((i & k) == 0);
                    const float va = v[i], vb = v[p];
                    const int   ia = ix[i], ib = ix[p];
                    const bool aFirst = (va > vb) || (va == vb && ia < ib);
                    if (aFirst != desc) {
                        v[i] = vb; v[p] = va; ix[i] = ib; ix[p] = ia;
                    }
                }
            }
            __syncthreads();
        }
    }
    for (int i = threadIdx.x; i < KTOK; i += 1024)
        idxout[(size_t)b * KTOK + i] = ix[i];
}

// ---------------------------------------------------------------------------
// Kernel 3a: zero mean/scale accumulators
// ---------------------------------------------------------------------------
__global__ void zero_kernel(float* __restrict__ p, int n)
{
    const int i = blockIdx.x * 256 + threadIdx.x;
    if (i < n) p[i] = 0.f;
}

// ---------------------------------------------------------------------------
// Kernel 3b: gathered mean & max|.| per (batch, dim), float4-vectorized.
// grid = (BATCH, 1, 8 token-splits); block = 256 threads covers D = 256*4.
// maxabs via u32 atomicMax on the bit pattern (valid: values >= 0).
// ---------------------------------------------------------------------------
__global__ __launch_bounds__(256)
void reduce_kernel(const float* __restrict__ x, const int* __restrict__ idx,
                   float* __restrict__ meanv, float* __restrict__ scalev)
{
    const int b  = blockIdx.x;
    const int d4 = threadIdx.x;              // float4 index, d = d4*4
    const int per = KTOK / 8;
    const int* ib = idx + (size_t)b * KTOK + blockIdx.z * per;

    float4 s = {0.f, 0.f, 0.f, 0.f};
    float4 m = {0.f, 0.f, 0.f, 0.f};
    for (int t = 0; t < per; ++t) {
        const int row = ib[t];
        const float4 val = *(const float4*)(x + ((size_t)b * SEQV + row) * DIMV + d4 * 4);
        s.x += val.x; s.y += val.y; s.z += val.z; s.w += val.w;
        m.x = fmaxf(m.x, fabsf(val.x)); m.y = fmaxf(m.y, fabsf(val.y));
        m.z = fmaxf(m.z, fabsf(val.z)); m.w = fmaxf(m.w, fabsf(val.w));
    }
    float* mp = meanv + (size_t)b * DIMV + d4 * 4;
    atomicAdd(mp + 0, s.x * (1.f / KTOK));
    atomicAdd(mp + 1, s.y * (1.f / KTOK));
    atomicAdd(mp + 2, s.z * (1.f / KTOK));
    atomicAdd(mp + 3, s.w * (1.f / KTOK));
    unsigned int* sp = (unsigned int*)(scalev + (size_t)b * DIMV + d4 * 4);
    atomicMax(sp + 0, __float_as_uint(m.x));
    atomicMax(sp + 1, __float_as_uint(m.y));
    atomicMax(sp + 2, __float_as_uint(m.z));
    atomicMax(sp + 3, __float_as_uint(m.w));
}

// ---------------------------------------------------------------------------
// Kernel 4: quant net -> num_bits per batch (argmax of logits; softmax monotone)
// ---------------------------------------------------------------------------
__global__ __launch_bounds__(256)
void qnet_kernel(const float* __restrict__ meanv, const float* __restrict__ wq1,
                 const float* __restrict__ bq1,   const float* __restrict__ wq2,
                 const float* __restrict__ bq2,   float* __restrict__ bitsArr)
{
    __shared__ float Hs[QHID];
    __shared__ float logits[8];
    const int b = blockIdx.x, j = threadIdx.x;

    float acc = bq1[j];
    const float* mb = meanv + (size_t)b * DIMV;
    for (int k = 0; k < DIMV; ++k) acc += mb[k] * wq1[(size_t)k * QHID + j];
    Hs[j] = fmaxf(acc, 0.f);
    __syncthreads();

    if (j < 8) {
        float l = bq2[j];
        for (int h = 0; h < QHID; ++h) l += Hs[h] * wq2[h * 8 + j];
        logits[j] = l;
    }
    __syncthreads();

    if (j == 0) {
        int best = 0; float bv = logits[0];
        for (int t = 1; t < 8; ++t) if (logits[t] > bv) { bv = logits[t]; best = t; }
        bitsArr[b] = (float)(best + 1);
    }
}

// ---------------------------------------------------------------------------
// Kernel 5: quantize selected tokens, float4-vectorized (b128 in/out).
// round == rintf (RTE) matches jnp.round.
// ---------------------------------------------------------------------------
__global__ __launch_bounds__(256)
void quant_kernel(const float* __restrict__ x, const int* __restrict__ idx,
                  const float* __restrict__ scalev, const float* __restrict__ bitsArr,
                  float* __restrict__ out)
{
    const size_t gid = (size_t)blockIdx.x * 256 + threadIdx.x;   // float4 units
    const size_t total4 = (size_t)BATCHV * KTOK * (DIMV / 4);
    if (gid >= total4) return;

    const int d4 = (int)(gid % (DIMV / 4));
    const size_t bk = gid / (DIMV / 4);
    const int k  = (int)(bk % KTOK);
    const int b  = (int)(bk / KTOK);

    const int row = idx[(size_t)b * KTOK + k];
    const float4 val = *(const float4*)(x + ((size_t)b * SEQV + row) * DIMV + d4 * 4);
    const float4 sc  = *(const float4*)(scalev + (size_t)b * DIMV + d4 * 4);
    const float inv2b = exp2f(-bitsArr[b]);    // 1 / 2^bits

    float4 o;
    const float sx = sc.x * inv2b, sy = sc.y * inv2b, sz = sc.z * inv2b, sw = sc.w * inv2b;
    o.x = rintf(val.x / sx) * sx;
    o.y = rintf(val.y / sy) * sy;
    o.z = rintf(val.z / sz) * sz;
    o.w = rintf(val.w / sw) * sw;
    *(float4*)(out + gid * 4) = o;
}

// ---------------------------------------------------------------------------
// Host launcher
// ---------------------------------------------------------------------------
extern "C" void kernel_launch(void* const* d_in, const int* in_sizes, int n_in,
                              void* d_out, int out_size, void* d_ws, size_t ws_size,
                              hipStream_t stream) {
    const float* x     = (const float*)d_in[0];
    const float* w1    = (const float*)d_in[1];
    const float* b1    = (const float*)d_in[2];
    const float* g1    = (const float*)d_in[3];
    const float* beta1 = (const float*)d_in[4];
    const float* w2    = (const float*)d_in[5];
    const float* b2    = (const float*)d_in[6];
    const float* wq1   = (const float*)d_in[7];
    const float* bq1   = (const float*)d_in[8];
    const float* wq2   = (const float*)d_in[9];
    const float* bq2   = (const float*)d_in[10];
    float* out = (float*)d_out;

    // workspace layout (w1sw first: 32B-aligned fragment base)
    float*    ws      = (float*)d_ws;
    _Float16* w1sw    = (_Float16*)ws;                         // 512K f16 = 1MB
    float*    imp     = ws + (DIMV * HDIM / 2);                // B*S  = 16384 f32
    int*      idxsel  = (int*)(imp + BATCHV * SEQV);           // B*K  =  8192 i32
    float*    meanv   = imp + BATCHV * SEQV + BATCHV * KTOK;   // B*D  =  4096 f32
    float*    scalev  = meanv + BATCHV * DIMV;                 // B*D  =  4096 f32
    float*    bitsArr = scalev + BATCHV * DIMV;                // B    =     4 f32

    // 0) swizzle w1 into WMMA B-fragment order (f16, L2-resident)
    w1_swizzle_kernel<<<(DIMV * HDIM) / 256, 256, 0, stream>>>(w1, w1sw);

    // 1) importance scores (WMMA GEMM + LN + sigmoid), 16 tokens/block
    scorer_kernel<<<BATCHV * (SEQV / 16), 256, 0, stream>>>(
        x, w1sw, b1, g1, beta1, w2, b2, imp);

    // 2) descending top-k indices per batch
    topk_kernel<<<BATCHV, 1024, 0, stream>>>(imp, idxsel);

    // 3) zero accumulators, then gathered mean / maxabs per dim
    zero_kernel<<<(2 * BATCHV * DIMV + 255) / 256, 256, 0, stream>>>(meanv, 2 * BATCHV * DIMV);
    {
        dim3 g(BATCHV, 1, 8);
        reduce_kernel<<<g, 256, 0, stream>>>(x, idxsel, meanv, scalev);
    }

    // 4) adaptive bit selection
    qnet_kernel<<<BATCHV, QHID, 0, stream>>>(meanv, wq1, bq1, wq2, bq2, bitsArr);

    // 5) quantize + write output (float4 granularity)
    {
        const size_t total4 = (size_t)BATCHV * KTOK * (DIMV / 4);
        quant_kernel<<<(unsigned)((total4 + 255) / 256), 256, 0, stream>>>(
            x, idxsel, scalev, bitsArr, out);
    }
}
